// NeSyBase_78469052498180
// MI455X (gfx1250) — compile-verified, matrix operand order checked by reading, not run on hardware
//
#include <hip/hip_runtime.h>
#include <hip/hip_bf16.h>
#include <math.h>

// Problem constants (from reference): B=8, K=4, N=16, S=64, H=512
#define BB   8
#define KK   4
#define NN   16
#define SS   64
#define HH   512
#define F2   1024              // 2H
#define ROWS 512               // B*K*N == B*S == 512 rows for both GEMMs
#define KDIM 1024              // GEMM reduction dim (2H)

typedef _Float16 h8   __attribute__((ext_vector_type(8)));
typedef _Float16 v16h __attribute__((ext_vector_type(16)));
typedef float    v8f  __attribute__((ext_vector_type(8)));

// ---------------------------------------------------------------------------
// 1) Convert f32 inputs to f16 operands. W1 halves are stored TRANSPOSED
//    ([h][k]) so that the WMMA B-fragment per-lane K-slices are contiguous.
// ---------------------------------------------------------------------------
__global__ void convert_kernel(const float* __restrict__ text,  // [512][1024]
                               const float* __restrict__ vid,   // [512][1024]
                               const float* __restrict__ W1,    // [2048][512]
                               _Float16* __restrict__ At,       // [512][1024]
                               _Float16* __restrict__ Av,       // [512][1024]
                               _Float16* __restrict__ Wt,       // [512][1024] = W1[:1024,:]^T
                               _Float16* __restrict__ Wb) {     // [512][1024] = W1[1024:,:]^T
  int idx    = blockIdx.x * blockDim.x + threadIdx.x;
  int region = idx >> 19;              // each region has 2^19 = 524288 elements
  int r      = idx & ((1 << 19) - 1);
  if (region == 0) {
    At[r] = (_Float16)text[r];
  } else if (region == 1) {
    Av[r] = (_Float16)vid[r];
  } else {
    int h = r >> 10;                   // 0..511
    int f = r & 1023;                  // 0..1023
    if (region == 2) Wt[r] = (_Float16)W1[(size_t)f * HH + h];
    else             Wb[r] = (_Float16)W1[(size_t)(F2 + f) * HH + h];
  }
}

// ---------------------------------------------------------------------------
// 2) 512x512x1024 GEMM: C = A * B, B given transposed as Bt[n][k].
//    One wave32 per 16x16 output tile; K-loop of v_wmma_f32_16x16x32_f16.
//    Fragment loads are 2 x b128 per operand per step (16B-aligned).
// ---------------------------------------------------------------------------
__global__ void gemm16_wmma(const _Float16* __restrict__ A,   // [512][1024]
                            const _Float16* __restrict__ Bt,  // [512][1024]
                            float* __restrict__ C) {          // [512][512]
  const int M = ROWS, Nn = HH, Kd = KDIM;
  int wid  = (blockIdx.x * blockDim.x + threadIdx.x) >> 5;
  int lane = threadIdx.x & 31;
  int tilesN = Nn / 16;
  int tm = wid / tilesN;
  int tn = wid % tilesN;
  if (tm >= M / 16) return;

  // 16-bit A-matrix 16x32 layout (wave32):
  //   lanes 0-15: M=lane, halves 0..7 -> K=kb+0..7, halves 8..15 -> K=kb+16..23
  //   lanes 16-31: same M, K offset +8
  int mrow = tm * 16 + (lane & 15);
  int ncol = tn * 16 + (lane & 15);
  int koff = (lane >> 4) * 8;
  const _Float16* arow = A  + (size_t)mrow * Kd + koff;
  const _Float16* brow = Bt + (size_t)ncol * Kd + koff;

  v8f acc = {};
  for (int kb = 0; kb < Kd; kb += 32) {
    h8 alo = *(const h8*)(arow + kb);
    h8 ahi = *(const h8*)(arow + kb + 16);
    h8 blo = *(const h8*)(brow + kb);
    h8 bhi = *(const h8*)(brow + kb + 16);
    v16h af, bf;
#pragma unroll
    for (int i = 0; i < 8; ++i) {
      af[i] = alo[i]; af[i + 8] = ahi[i];
      bf[i] = blo[i]; bf[i + 8] = bhi[i];
    }
    // 8 args: (neg_a, A, neg_b, B, c_mod, C, reuse_a, reuse_b)
    acc = __builtin_amdgcn_wmma_f32_16x16x32_f16(
        false, af, false, bf, (short)0, acc, false, false);
  }

  // 32-bit C/D 16x16 layout: VGPR r -> row r (lanes 0-15) / row r+8 (lanes 16-31)
  int crow0 = tm * 16 + (lane >> 4) * 8;
  int ccol  = tn * 16 + (lane & 15);
#pragma unroll
  for (int r = 0; r < 8; ++r)
    C[(size_t)(crow0 + r) * Nn + ccol] = acc[r];
}

// ---------------------------------------------------------------------------
// 3) logits[b,k,n,s] = b2 + sum_h relu(T[bkn,h] + V[b*S+s,h] + b1[h]) * w2[h]
//    One wave32 per output element; each lane owns 16 channels (4 x float4),
//    wave reduction via shfl_xor. T/V are L2-resident (1 MB each).
// ---------------------------------------------------------------------------
__global__ void logits_kernel(const float* __restrict__ T,   // [512][512]
                              const float* __restrict__ V,   // [512][512]
                              const float* __restrict__ b1,  // [512]
                              const float* __restrict__ w2,  // [512]
                              const float* __restrict__ b2,  // [1]
                              float* __restrict__ logits) {  // [B*K*N*S]
  int gwid = (blockIdx.x * blockDim.x + threadIdx.x) >> 5;
  int lane = threadIdx.x & 31;
  if (gwid >= BB * KK * NN * SS) return;
  int s   = gwid & (SS - 1);
  int bkn = gwid / SS;
  int b   = bkn / (KK * NN);

  const float4* trow = (const float4*)(T + (size_t)bkn * HH) + lane * 4;
  const float4* vrow = (const float4*)(V + (size_t)(b * SS + s) * HH) + lane * 4;
  const float4* b1v  = (const float4*)b1 + lane * 4;
  const float4* w2v  = (const float4*)w2 + lane * 4;

  float sum = 0.f;
#pragma unroll
  for (int i = 0; i < 4; ++i) {
    float4 t = trow[i], v = vrow[i], bb = b1v[i], w = w2v[i];
    float x;
    x = t.x + v.x + bb.x; sum += fmaxf(x, 0.f) * w.x;
    x = t.y + v.y + bb.y; sum += fmaxf(x, 0.f) * w.y;
    x = t.z + v.z + bb.z; sum += fmaxf(x, 0.f) * w.z;
    x = t.w + v.w + bb.w; sum += fmaxf(x, 0.f) * w.w;
  }
#pragma unroll
  for (int off = 16; off; off >>= 1) sum += __shfl_xor(sum, off, 32);
  if (lane == 0) logits[gwid] = sum + b2[0];
}

// ---------------------------------------------------------------------------
// 4) Banded Viterbi DP + traceback + per-b argmax over k + outputs.
//    One thread per (b,k) (32 total, tiny 16x64 DP each), then 8 threads
//    finalize. Outputs: [0..7]=probs, [8..15]=labels, [16..143]=align (float).
// ---------------------------------------------------------------------------
__device__ __forceinline__ float log_sigmoid(float x) {
  return (x >= 0.f) ? -log1pf(expf(-x)) : (x - log1pf(expf(x)));
}

__global__ void dp_kernel(const float* __restrict__ logits,  // [B*K][N][S]
                          const float* __restrict__ labels,  // [B]
                          float* __restrict__ out) {         // [144]
  __shared__ float sBest[BB * KK];
  __shared__ float sAgg[BB * KK];
  __shared__ int   sAlign[BB * KK][NN];
  const float NEG = -100.0f;
  int t = threadIdx.x;

  if (t < BB * KK) {
    const float* Lg = logits + (size_t)t * NN * SS;
    float arr[SS], c[SS], row[SS];
    unsigned long long choice[NN];
    for (int s = 0; s < SS; ++s) arr[s] = NEG;

    for (int n = NN - 1; n >= 0; --n) {
      int lo = n, hi = SS - NN + n;
      for (int s = 0; s < SS; ++s) {
        float ls = log_sigmoid(Lg[n * SS + s]);
        c[s] = (n == NN - 1) ? ls : ls + arr[s];
      }
      // suffix running-max over the band
      float run = -__builtin_inff();
      for (int s = SS - 1; s >= 0; --s) {
        bool ib = (s >= lo) && (s <= hi);
        if (ib) run = fmaxf(run, c[s]);
        row[s] = ib ? fmaxf(run, NEG) : NEG;
      }
      unsigned long long bits = 0ull;
      for (int s = 0; s < SS; ++s) {
        bool ib = (s >= lo) && (s <= hi);
        float rn = (s < SS - 1) ? row[s + 1] : -1e30f;
        if (ib && (c[s] >= rn)) bits |= (1ull << s);
      }
      choice[n] = bits;
      for (int s = 0; s < SS; ++s) arr[s] = row[s];
    }
    float best = arr[0];

    // traceback (faithful to reference scan)
    int node = 0, seg = 0;
    bool done = false;
    float agg = 0.f;
    int align[NN];
    for (int n = 0; n < NN; ++n) align[n] = 0;
    for (int step = 0; step < NN + SS; ++step) {
      bool took = (!done) && ((choice[node] >> seg) & 1ull);
      if (took) { agg += Lg[node * SS + seg]; align[node] = seg; }
      done = done || (took && node == NN - 1);
      if (took && !done) node += 1;
      if (!done) { int ns2 = seg + 1; seg = (ns2 > SS - 1) ? (SS - 1) : ns2; }
    }

    sBest[t] = best;
    sAgg[t]  = agg;
    for (int n = 0; n < NN; ++n) sAlign[t][n] = align[n];
  }
  __syncthreads();

  if (t < BB) {
    int kb = 0;
    float bv = sBest[t * KK];
    for (int k = 1; k < KK; ++k) {
      float v = sBest[t * KK + k];
      if (v > bv) { bv = v; kb = k; }   // strict > keeps first max (argmax tie rule)
    }
    float agg = sAgg[t * KK + kb];
    out[t]      = 1.0f / (1.0f + expf(-agg));   // probs
    out[BB + t] = labels[t];                    // labels pass-through
    for (int n = 0; n < NN; ++n)
      out[2 * BB + t * NN + n] = (float)sAlign[t * KK + kb][n];
  }
}

// ---------------------------------------------------------------------------
// Launch
// ---------------------------------------------------------------------------
extern "C" void kernel_launch(void* const* d_in, const int* in_sizes, int n_in,
                              void* d_out, int out_size, void* d_ws, size_t ws_size,
                              hipStream_t stream) {
  const float* text   = (const float*)d_in[0];  // [8,4,16,1024]
  const float* vid    = (const float*)d_in[1];  // [8,64,1024]
  const float* labels = (const float*)d_in[2];  // [8]
  const float* W1     = (const float*)d_in[3];  // [2048,512]
  const float* b1     = (const float*)d_in[4];  // [512]
  const float* w2     = (const float*)d_in[5];  // [512]
  const float* b2     = (const float*)d_in[6];  // [1]
  float* out = (float*)d_out;

  char* ws = (char*)d_ws;
  const size_t MB = 1024 * 1024;
  _Float16* At = (_Float16*)(ws + 0 * MB);  // 1 MB
  _Float16* Av = (_Float16*)(ws + 1 * MB);  // 1 MB
  _Float16* Wt = (_Float16*)(ws + 2 * MB);  // 1 MB
  _Float16* Wb = (_Float16*)(ws + 3 * MB);  // 1 MB
  float*    T  = (float*)(ws + 4 * MB);     // 1 MB
  float*    V  = (float*)(ws + 5 * MB);     // 1 MB
  float*    Lg = (float*)(ws + 6 * MB);     // 128 KB

  // 4 regions x 524288 elements, 256 threads/block
  convert_kernel<<<(4 * 524288) / 256, 256, 0, stream>>>(text, vid, W1, At, Av, Wt, Wb);

  // 1024 tiles per GEMM, 8 waves/block -> 128 blocks
  gemm16_wmma<<<128, 256, 0, stream>>>(At, Wt, T);
  gemm16_wmma<<<128, 256, 0, stream>>>(Av, Wb, V);

  // 32768 output elements, 1 wave each, 8 waves/block -> 4096 blocks
  logits_kernel<<<4096, 256, 0, stream>>>(T, V, b1, w2, b2, Lg);

  dp_kernel<<<1, 32, 0, stream>>>(Lg, labels, out);
}
